// GCN3D_61383672594536
// MI455X (gfx1250) — matrix operand order repaired and checked by reading, not compiled, not run on hardware
//
#include <hip/hip_runtime.h>

typedef __attribute__((ext_vector_type(16))) _Float16 v16h;
typedef __attribute__((ext_vector_type(8)))  float    v8f;
typedef _Float16 half_t;

#define BS   4
#define NV   4096
#define NV1  1024
#define NV2  256
#define KNB  32
#define EPSF 1e-12f
#define FINF 3.402823466e38f

// ---------------------------------------------------------------------------
// KNN: one thread per query; insertion-sorted top-k (k <= 32), excludes self.
// ---------------------------------------------------------------------------
__global__ void knn_kernel(const float* __restrict__ pts, const int* __restrict__ sidx,
                           int Nc, int Nq, int k, int* __restrict__ out)
{
    int t = blockIdx.x * blockDim.x + threadIdx.x;
    if (t >= BS * Nq) return;
    int b = t / Nq, q = t - b * Nq;
    int self = sidx ? sidx[q] : q;
    const float* P = pts + (size_t)b * Nc * 3;
    float qx = P[3*self+0], qy = P[3*self+1], qz = P[3*self+2];
    float bd[KNB]; int bi[KNB];
    #pragma unroll
    for (int i = 0; i < KNB; i++) { bd[i] = FINF; bi[i] = 0; }
    for (int j = 0; j < Nc; j++) {
        if (j == self) continue;
        float dx = P[3*j+0]-qx, dy = P[3*j+1]-qy, dz = P[3*j+2]-qz;
        float d = dx*dx + dy*dy + dz*dz;
        if (d < bd[k-1]) {
            int p = k - 1;
            while (p > 0 && bd[p-1] > d) { bd[p] = bd[p-1]; bi[p] = bi[p-1]; p--; }
            bd[p] = d; bi[p] = j;
        }
    }
    int* o = out + ((size_t)b * Nq + q) * k;
    for (int i = 0; i < k; i++) o[i] = bi[i];
}

// ---------------------------------------------------------------------------
// conv_surface: fm0[b,v,o] = relu(max_n relu(dot(nbr_dir_n, dir_col_o)))
// ---------------------------------------------------------------------------
__global__ void conv_surface_kernel(const float* __restrict__ verts, const int* __restrict__ nb,
                                    const float* __restrict__ dir, half_t* __restrict__ out)
{
    int bv = blockIdx.x;
    int b = bv / NV, v = bv - b * NV;
    __shared__ float sd[KNB][3];
    const float* P = verts + (size_t)b * NV * 3;
    float cx = P[3*v], cy = P[3*v+1], cz = P[3*v+2];
    int tid = threadIdx.x;
    if (tid < KNB) {
        int j = nb[((size_t)b * NV + v) * KNB + tid];
        float dx = P[3*j]-cx, dy = P[3*j+1]-cy, dz = P[3*j+2]-cz;
        float inv = 1.0f / fmaxf(sqrtf(dx*dx+dy*dy+dz*dz), EPSF);
        sd[tid][0] = dx*inv; sd[tid][1] = dy*inv; sd[tid][2] = dz*inv;
    }
    __syncthreads();
    int o = tid;
    float d0 = dir[o], d1 = dir[128+o], d2 = dir[256+o];
    float inv = 1.0f / fmaxf(sqrtf(d0*d0+d1*d1+d2*d2), EPSF);
    d0 *= inv; d1 *= inv; d2 *= inv;
    float m = 0.0f;
    #pragma unroll
    for (int n = 0; n < KNB; n++)
        m = fmaxf(m, sd[n][0]*d0 + sd[n][1]*d1 + sd[n][2]*d2);
    out[((size_t)b * NV + v) * 128 + o] = (half_t)m;
}

// ---------------------------------------------------------------------------
// conv_layer combine: out[b,v,o] = [relu]( f[b,v,o] +
//     max_n( relu(dot(nbr_dir, dir_col_o)) * f[b, nb[b,v,n], OC+o] ) )
// ---------------------------------------------------------------------------
__global__ void conv_combine_kernel(const float* __restrict__ pts, int Np,
                                    const int* __restrict__ nb,
                                    const float* __restrict__ f,
                                    const float* __restrict__ dirs,
                                    int OC, int do_relu, half_t* __restrict__ out)
{
    int bv = blockIdx.x;
    int b = bv / Np, v = bv - b * Np;
    __shared__ float sdir[KNB*3];
    __shared__ int   snb[KNB];
    const float* P = pts + (size_t)b * Np * 3;
    float cx = P[3*v], cy = P[3*v+1], cz = P[3*v+2];
    int tid = threadIdx.x;
    if (tid < KNB) {
        int j = nb[((size_t)b * Np + v) * KNB + tid];
        snb[tid] = j;
        float dx = P[3*j]-cx, dy = P[3*j+1]-cy, dz = P[3*j+2]-cz;
        float inv = 1.0f / fmaxf(sqrtf(dx*dx+dy*dy+dz*dz), EPSF);
        sdir[3*tid+0] = dx*inv; sdir[3*tid+1] = dy*inv; sdir[3*tid+2] = dz*inv;
    }
    __syncthreads();
    int o = tid;
    float d0 = dirs[o], d1 = dirs[OC+o], d2 = dirs[2*OC+o];
    float inv = 1.0f / fmaxf(sqrtf(d0*d0+d1*d1+d2*d2), EPSF);
    d0 *= inv; d1 *= inv; d2 *= inv;
    const float* Fb = f + (size_t)b * Np * (2*OC);
    float best = -FINF;
    #pragma unroll 4
    for (int n = 0; n < KNB; n++) {
        float th = fmaxf(0.0f, sdir[3*n]*d0 + sdir[3*n+1]*d1 + sdir[3*n+2]*d2);
        float s = Fb[(size_t)snb[n] * (2*OC) + OC + o];
        best = fmaxf(best, th * s);
    }
    float r = Fb[(size_t)v * (2*OC) + o] + best;
    if (do_relu) r = fmaxf(r, 0.0f);
    out[((size_t)b * Np + v) * OC + o] = (half_t)r;
}

// ---------------------------------------------------------------------------
// WMMA fragment loaders (layouts per CDNA5 ISA 7.12.2).
// A 16x32 f16: p = &A[row*K + kb + 8*hi]; elems 0-7 at +0, elems 8-15 at +16.
// B 32x16 f16: p = &Bt[col*K + kb + 8*hi]; elems 0-7 at +0, elems 8-15 at +16.
// ---------------------------------------------------------------------------
static __device__ __forceinline__ v16h load_a_frag(const half_t* __restrict__ p)
{
    v16h r;
    #pragma unroll
    for (int e = 0; e < 8; e++) r[e]   = p[e];
    #pragma unroll
    for (int e = 0; e < 8; e++) r[8+e] = p[16 + e];
    return r;
}
static __device__ __forceinline__ v16h load_b_frag(const half_t* __restrict__ p)
{
    v16h r;
    #pragma unroll
    for (int e = 0; e < 16; e++) r[e] = p[(e & 7) + ((e >> 3) << 4)];
    return r;
}

// ---------------------------------------------------------------------------
// WMMA GEMM: C[M x Nact] = A[M x K](f16) * Bt[Npad x K](f16) + bias.
// Block = 256 threads = 8 waves; wave owns a 32x64 output tile:
// 2 A fragments x 4 B fragments -> 8 accumulators (8 WMMA / 12 b128-loads per
// 32-wide k-step). Software-pipelined: next k-step's fragments are loaded
// before the current step's 8 WMMAs (last step peeled), hiding load latency
// behind the matrix pipe. Block covers 256 rows; M % 256 == 0, Npad % 64 == 0.
// Output: f32 to Cf if non-null, else f16 to Ch.
// ---------------------------------------------------------------------------
__global__ void __launch_bounds__(256)
wmma_gemm_kernel(const half_t* __restrict__ A, const half_t* __restrict__ Bt,
                 const float* __restrict__ bias,
                 float* __restrict__ Cf, half_t* __restrict__ Ch,
                 int M, int K, int Nact, int ldc, int do_relu)
{
    int lane = threadIdx.x & 31;
    int wave = threadIdx.x >> 5;
    int m_tile = blockIdx.y * 256 + wave * 32;
    int n_base = blockIdx.x * 64;
    int hi = lane >> 4;          // 0: lanes 0-15, 1: lanes 16-31
    int lr = lane & 15;
    if (m_tile >= M) return;

    const half_t* Ap0 = A  + (size_t)(m_tile + lr)      * K + 8 * hi;
    const half_t* Ap1 = A  + (size_t)(m_tile + 16 + lr) * K + 8 * hi;
    const half_t* Bp  = Bt + (size_t)(n_base + lr)      * K + 8 * hi;

    v8f acc[8] = {};   // acc[ai*4 + f]: rows m_tile+16*ai.., cols n_base+16*f..

    v16h a0 = load_a_frag(Ap0);
    v16h a1 = load_a_frag(Ap1);
    v16h b0 = load_b_frag(Bp);
    v16h b1 = load_b_frag(Bp + (size_t)16 * K);
    v16h b2 = load_b_frag(Bp + (size_t)32 * K);
    v16h b3 = load_b_frag(Bp + (size_t)48 * K);

    for (int kb = 32; kb < K; kb += 32) {
        v16h na0 = load_a_frag(Ap0 + kb);
        v16h na1 = load_a_frag(Ap1 + kb);
        v16h nb0 = load_b_frag(Bp + kb);
        v16h nb1 = load_b_frag(Bp + (size_t)16 * K + kb);
        v16h nb2 = load_b_frag(Bp + (size_t)32 * K + kb);
        v16h nb3 = load_b_frag(Bp + (size_t)48 * K + kb);

        acc[0] = __builtin_amdgcn_wmma_f32_16x16x32_f16(false, a0, false, b0, (short)0, acc[0], false, false);
        acc[1] = __builtin_amdgcn_wmma_f32_16x16x32_f16(false, a0, false, b1, (short)0, acc[1], false, false);
        acc[2] = __builtin_amdgcn_wmma_f32_16x16x32_f16(false, a0, false, b2, (short)0, acc[2], false, false);
        acc[3] = __builtin_amdgcn_wmma_f32_16x16x32_f16(false, a0, false, b3, (short)0, acc[3], false, false);
        acc[4] = __builtin_amdgcn_wmma_f32_16x16x32_f16(false, a1, false, b0, (short)0, acc[4], false, false);
        acc[5] = __builtin_amdgcn_wmma_f32_16x16x32_f16(false, a1, false, b1, (short)0, acc[5], false, false);
        acc[6] = __builtin_amdgcn_wmma_f32_16x16x32_f16(false, a1, false, b2, (short)0, acc[6], false, false);
        acc[7] = __builtin_amdgcn_wmma_f32_16x16x32_f16(false, a1, false, b3, (short)0, acc[7], false, false);

        a0 = na0; a1 = na1; b0 = nb0; b1 = nb1; b2 = nb2; b3 = nb3;
    }
    // peeled last k-step
    acc[0] = __builtin_amdgcn_wmma_f32_16x16x32_f16(false, a0, false, b0, (short)0, acc[0], false, false);
    acc[1] = __builtin_amdgcn_wmma_f32_16x16x32_f16(false, a0, false, b1, (short)0, acc[1], false, false);
    acc[2] = __builtin_amdgcn_wmma_f32_16x16x32_f16(false, a0, false, b2, (short)0, acc[2], false, false);
    acc[3] = __builtin_amdgcn_wmma_f32_16x16x32_f16(false, a0, false, b3, (short)0, acc[3], false, false);
    acc[4] = __builtin_amdgcn_wmma_f32_16x16x32_f16(false, a1, false, b0, (short)0, acc[4], false, false);
    acc[5] = __builtin_amdgcn_wmma_f32_16x16x32_f16(false, a1, false, b1, (short)0, acc[5], false, false);
    acc[6] = __builtin_amdgcn_wmma_f32_16x16x32_f16(false, a1, false, b2, (short)0, acc[6], false, false);
    acc[7] = __builtin_amdgcn_wmma_f32_16x16x32_f16(false, a1, false, b3, (short)0, acc[7], false, false);

    #pragma unroll
    for (int ai = 0; ai < 2; ai++) {
        #pragma unroll
        for (int f = 0; f < 4; f++) {
            int n = n_base + f * 16 + lr;
            if (n >= Nact) continue;
            float bv = bias ? bias[n] : 0.0f;
            #pragma unroll
            for (int r = 0; r < 8; r++) {       // C/D: M = r + 8*hi, N = lr
                float val = acc[ai*4 + f][r] + bv;
                if (do_relu) val = fmaxf(val, 0.0f);
                size_t off = (size_t)(m_tile + ai*16 + r + 8*hi) * ldc + n;
                if (Cf) Cf[off] = val;
                else    Ch[off] = (half_t)val;
            }
        }
    }
}

// Weight conversion: W[K][Nw] (f32) -> Bt[Npad][K] (f16, transposed, zero-pad)
__global__ void wcvt_kn_to_nk(const float* __restrict__ W, int K, int Nw, int Npad,
                              half_t* __restrict__ Bt)
{
    int t = blockIdx.x * blockDim.x + threadIdx.x;
    if (t >= Npad * K) return;
    int n = t / K, k = t - n * K;
    Bt[t] = (n < Nw) ? (half_t)W[(size_t)k * Nw + n] : (half_t)0.0f;
}

// Weight conversion: W[Nw][K] (f32, out-major) -> Bt[Npad][K] (f16, zero-pad)
__global__ void wcvt_nk(const float* __restrict__ W, int K, int Nw, int Npad,
                        half_t* __restrict__ Bt)
{
    int t = blockIdx.x * blockDim.x + threadIdx.x;
    if (t >= Npad * K) return;
    int n = t / K, k = t - n * K;
    Bt[t] = (n < Nw) ? (half_t)W[(size_t)n * K + k] : (half_t)0.0f;
}

// pooled[b,s,c] = max over 4 knn neighbors of fm[b, nb4[b,s,t], c]  (f16 in/out)
__global__ void pool_gather_kernel(const half_t* __restrict__ fm, const int* __restrict__ nb4,
                                   int Np, int C, int Ns, half_t* __restrict__ out)
{
    int t = blockIdx.x * blockDim.x + threadIdx.x;
    if (t >= BS * Ns * C) return;
    int c = t % C; int bs = t / C; int b = bs / Ns; int s = bs - b * Ns;
    const int* nn = nb4 + ((size_t)b * Ns + s) * 4;
    const half_t* F = fm + (size_t)b * Np * C;
    float m = -FINF;
    #pragma unroll
    for (int i = 0; i < 4; i++) m = fmaxf(m, (float)F[(size_t)nn[i] * C + c]);
    out[t] = (half_t)m;
}

__global__ void sample_verts_kernel(const float* __restrict__ pts, const int* __restrict__ sidx,
                                    int Np, int Ns, float* __restrict__ out)
{
    int t = blockIdx.x * blockDim.x + threadIdx.x;
    if (t >= BS * Ns) return;
    int b = t / Ns, s = t - b * Ns;
    int j = sidx[s];
    const float* src = pts + ((size_t)b * Np + j) * 3;
    float* dst = out + (size_t)t * 3;
    dst[0] = src[0]; dst[1] = src[1]; dst[2] = src[2];
}

__global__ void nearest_kernel(const float* __restrict__ tgt, const float* __restrict__ src,
                               int Nsrc, int* __restrict__ out)
{
    int t = blockIdx.x * blockDim.x + threadIdx.x;
    if (t >= BS * NV) return;
    int b = t / NV;
    const float* q = tgt + (size_t)t * 3;
    float qx = q[0], qy = q[1], qz = q[2];
    const float* S = src + (size_t)b * Nsrc * 3;
    float bestd = FINF; int besti = 0;
    for (int j = 0; j < Nsrc; j++) {
        float dx = S[3*j]-qx, dy = S[3*j+1]-qy, dz = S[3*j+2]-qz;
        float d = dx*dx + dy*dy + dz*dz;
        if (d < bestd) { bestd = d; besti = j; }
    }
    out[t] = besti;
}

__global__ void rowmax_kernel(const half_t* __restrict__ fm4, half_t* __restrict__ fg)
{
    int b = blockIdx.x, c = threadIdx.x;   // 512 threads
    float m = -FINF;
    for (int v = 0; v < NV2; v++)
        m = fmaxf(m, (float)fm4[((size_t)b * NV2 + v) * 512 + c]);
    fg[(size_t)b * 512 + c] = (half_t)m;
}

__global__ void fuse_kernel(const half_t* __restrict__ fm0, const half_t* __restrict__ fm1,
                            const half_t* __restrict__ fm2, const half_t* __restrict__ fm3,
                            const half_t* __restrict__ fm4, const half_t* __restrict__ fg,
                            const int* __restrict__ near1, const int* __restrict__ near2,
                            half_t* __restrict__ fuse)
{
    size_t t = (size_t)blockIdx.x * blockDim.x + threadIdx.x;
    if (t >= (size_t)BS * NV * 1792) return;
    int c = (int)(t % 1792);
    size_t bv = t / 1792;
    int b = (int)(bv / NV);
    half_t val;
    if      (c < 128)  val = fm0[bv * 128 + c];
    else if (c < 256)  val = fm1[bv * 128 + (c - 128)];
    else if (c < 512)  val = fm2[((size_t)b * NV1 + near1[bv]) * 256 + (c - 256)];
    else if (c < 768)  val = fm3[((size_t)b * NV1 + near1[bv]) * 256 + (c - 512)];
    else if (c < 1280) val = fm4[((size_t)b * NV2 + near2[bv]) * 512 + (c - 768)];
    else               val = fg[(size_t)b * 512 + (c - 1280)];
    fuse[t] = val;
}

// ---------------------------------------------------------------------------
extern "C" void kernel_launch(void* const* d_in, const int* in_sizes, int n_in,
                              void* d_out, int out_size, void* d_ws, size_t ws_size,
                              hipStream_t stream)
{
    const float* vertices = (const float*)d_in[0];
    const int*   sidx1    = (const int*)  d_in[1];
    const int*   sidx2    = (const int*)  d_in[2];
    const float* dir0     = (const float*)d_in[3];
    const float* w1 = (const float*)d_in[4],  *b1 = (const float*)d_in[5],  *dd1 = (const float*)d_in[6];
    const float* w2 = (const float*)d_in[7],  *b2 = (const float*)d_in[8],  *dd2 = (const float*)d_in[9];
    const float* w3 = (const float*)d_in[10], *b3 = (const float*)d_in[11], *dd3 = (const float*)d_in[12];
    const float* w4 = (const float*)d_in[13], *b4 = (const float*)d_in[14], *dd4 = (const float*)d_in[15];
    const float* cw1 = (const float*)d_in[16], *cb1 = (const float*)d_in[17];
    const float* cw2 = (const float*)d_in[18], *cb2 = (const float*)d_in[19];
    const float* cw3 = (const float*)d_in[20], *cb3 = (const float*)d_in[21];
    (void)in_sizes; (void)n_in; (void)ws_size; (void)out_size;

    char* p = (char*)d_ws;
    auto alloc = [&](size_t bytes) -> char* {
        char* r = p; p += (bytes + 255) & ~(size_t)255; return r;
    };
    auto cdiv = [](long long a, long long b) -> int { return (int)((a + b - 1) / b); };

    int*    nb0   = (int*)   alloc((size_t)BS*NV*KNB*4);
    half_t* fm0   = (half_t*)alloc((size_t)BS*NV*128*2);
    float*  f1    = (float*) alloc((size_t)BS*NV*256*4);
    half_t* fm1   = (half_t*)alloc((size_t)BS*NV*128*2);
    int*    nb4a  = (int*)   alloc((size_t)BS*NV1*4*4);
    float*  vp1   = (float*) alloc((size_t)BS*NV1*3*4);
    half_t* fmp1  = (half_t*)alloc((size_t)BS*NV1*128*2);
    int*    nb1   = (int*)   alloc((size_t)BS*NV1*KNB*4);
    float*  f2    = (float*) alloc((size_t)BS*NV1*512*4);
    half_t* fm2   = (half_t*)alloc((size_t)BS*NV1*256*2);
    float*  f3    = (float*) alloc((size_t)BS*NV1*512*4);
    half_t* fm3   = (half_t*)alloc((size_t)BS*NV1*256*2);
    int*    nb4b  = (int*)   alloc((size_t)BS*NV2*4*4);
    float*  vp2   = (float*) alloc((size_t)BS*NV2*3*4);
    half_t* fmp2  = (half_t*)alloc((size_t)BS*NV2*256*2);
    int*    nb2   = (int*)   alloc((size_t)BS*NV2*KNB*4);
    float*  f4    = (float*) alloc((size_t)BS*NV2*1024*4);
    half_t* fm4   = (half_t*)alloc((size_t)BS*NV2*512*2);
    half_t* fg    = (half_t*)alloc((size_t)BS*512*2);
    int*    near1 = (int*)   alloc((size_t)BS*NV*4);
    int*    near2 = (int*)   alloc((size_t)BS*NV*4);
    half_t* fuse  = (half_t*)alloc((size_t)BS*NV*1792*2);
    half_t* h1    = (half_t*)alloc((size_t)BS*NV*512*2);
    half_t* h2    = (half_t*)alloc((size_t)BS*NV*512*2);
    half_t* w1t   = (half_t*)alloc((size_t)256*128*2);
    half_t* w2t   = (half_t*)alloc((size_t)512*128*2);
    half_t* w3t   = (half_t*)alloc((size_t)512*256*2);
    half_t* w4t   = (half_t*)alloc((size_t)1024*256*2);
    half_t* cw1t  = (half_t*)alloc((size_t)512*1792*2);
    half_t* cw2t  = (half_t*)alloc((size_t)512*512*2);
    half_t* cw3t  = (half_t*)alloc((size_t)64*512*2);   // 13 real rows, padded to 64

    // Weight conversion to transposed f16 (output-col-major, K contiguous)
    wcvt_kn_to_nk<<<cdiv(256*128,256),256,0,stream>>>(w1,128,256,256,w1t);
    wcvt_kn_to_nk<<<cdiv(512*128,256),256,0,stream>>>(w2,128,512,512,w2t);
    wcvt_kn_to_nk<<<cdiv(512*256,256),256,0,stream>>>(w3,256,512,512,w3t);
    wcvt_kn_to_nk<<<cdiv(1024*256,256),256,0,stream>>>(w4,256,1024,1024,w4t);
    wcvt_nk<<<cdiv(512*1792,256),256,0,stream>>>(cw1,1792,512,512,cw1t);
    wcvt_nk<<<cdiv(512*512,256),256,0,stream>>>(cw2,512,512,512,cw2t);
    wcvt_nk<<<cdiv(64*512,256),256,0,stream>>>(cw3,512,13,64,cw3t);

    // Stage 1 (N=4096)
    knn_kernel<<<cdiv(BS*NV,128),128,0,stream>>>(vertices,nullptr,NV,NV,KNB,nb0);
    conv_surface_kernel<<<BS*NV,128,0,stream>>>(vertices,nb0,dir0,fm0);
    wmma_gemm_kernel<<<dim3(256/64,(BS*NV)/256),256,0,stream>>>(fm0,w1t,b1,f1,nullptr,BS*NV,128,256,256,0);
    conv_combine_kernel<<<BS*NV,128,0,stream>>>(vertices,NV,nb0,f1,dd1,128,1,fm1);

    // Pool 1 -> 1024 points
    knn_kernel<<<cdiv(BS*NV1,128),128,0,stream>>>(vertices,sidx1,NV,NV1,4,nb4a);
    sample_verts_kernel<<<cdiv(BS*NV1,256),256,0,stream>>>(vertices,sidx1,NV,NV1,vp1);
    pool_gather_kernel<<<cdiv((long long)BS*NV1*128,256),256,0,stream>>>(fm1,nb4a,NV,128,NV1,fmp1);

    // Stage 2 (N=1024)
    knn_kernel<<<cdiv(BS*NV1,128),128,0,stream>>>(vp1,nullptr,NV1,NV1,KNB,nb1);
    wmma_gemm_kernel<<<dim3(512/64,(BS*NV1)/256),256,0,stream>>>(fmp1,w2t,b2,f2,nullptr,BS*NV1,128,512,512,0);
    conv_combine_kernel<<<BS*NV1,256,0,stream>>>(vp1,NV1,nb1,f2,dd2,256,1,fm2);
    wmma_gemm_kernel<<<dim3(512/64,(BS*NV1)/256),256,0,stream>>>(fm2,w3t,b3,f3,nullptr,BS*NV1,256,512,512,0);
    conv_combine_kernel<<<BS*NV1,256,0,stream>>>(vp1,NV1,nb1,f3,dd3,256,1,fm3);

    // Pool 2 -> 256 points
    knn_kernel<<<cdiv(BS*NV2,128),128,0,stream>>>(vp1,sidx2,NV1,NV2,4,nb4b);
    sample_verts_kernel<<<cdiv(BS*NV2,256),256,0,stream>>>(vp1,sidx2,NV1,NV2,vp2);
    pool_gather_kernel<<<cdiv((long long)BS*NV2*256,256),256,0,stream>>>(fm3,nb4b,NV1,256,NV2,fmp2);

    // Stage 3 (N=256)
    knn_kernel<<<cdiv(BS*NV2,128),128,0,stream>>>(vp2,nullptr,NV2,NV2,KNB,nb2);
    wmma_gemm_kernel<<<dim3(1024/64,(BS*NV2)/256),256,0,stream>>>(fmp2,w4t,b4,f4,nullptr,BS*NV2,256,1024,1024,0);
    conv_combine_kernel<<<BS*NV2,512,0,stream>>>(vp2,NV2,nb2,f4,dd4,512,0,fm4);

    // Global feature + upsampling indices
    rowmax_kernel<<<BS,512,0,stream>>>(fm4,fg);
    nearest_kernel<<<cdiv(BS*NV,128),128,0,stream>>>(vertices,vp1,NV1,near1);
    nearest_kernel<<<cdiv(BS*NV,128),128,0,stream>>>(vertices,vp2,NV2,near2);

    // Fusion + classifier MLP (h1/h2 stay f16; logits f32 to d_out)
    fuse_kernel<<<cdiv((long long)BS*NV*1792,256),256,0,stream>>>(fm0,fm1,fm2,fm3,fm4,fg,near1,near2,fuse);
    wmma_gemm_kernel<<<dim3(512/64,(BS*NV)/256),256,0,stream>>>(fuse,cw1t,cb1,nullptr,h1,BS*NV,1792,512,512,1);
    wmma_gemm_kernel<<<dim3(512/64,(BS*NV)/256),256,0,stream>>>(h1,cw2t,cb2,nullptr,h2,BS*NV,512,512,512,1);
    wmma_gemm_kernel<<<dim3(1,(BS*NV)/256),256,0,stream>>>(h2,cw3t,cb3,(float*)d_out,nullptr,BS*NV,512,13,13,0);
}
